// Attention__42623255446033
// MI455X (gfx1250) — matrix operand order repaired and checked by reading, not compiled, not run on hardware
//
#include <hip/hip_runtime.h>

// ---------------- CDNA5 vector types (plain clang ext-vectors, POD) --------
typedef __attribute__((ext_vector_type(16))) __bf16       v16bf;
typedef __attribute__((ext_vector_type(8)))  float        v8f;
typedef __attribute__((ext_vector_type(4)))  unsigned int u32x4;
typedef __attribute__((ext_vector_type(2)))  unsigned int u32x2;
typedef __attribute__((ext_vector_type(4)))  float        f32x4;

#define B_    2
#define H_    8
#define N_    2048
#define D_    128
#define E_    64
#define CHUNK 128
#define NC    (N_ / CHUNK)

// LDS pitches (bf16 elements); 136*2 = 272 B per row -> staggers banks
#define QP 136   // for 128-wide arrays (sQ, sK, sKT, sAm)
#define VP 136   // for 64-row x 128-col arrays (sVt, sSt)

// dynamic-LDS byte offsets (all 16B aligned)
#define OFF_Q   0
#define OFF_K   (OFF_Q  + 128 * QP * 2)
#define OFF_KT  (OFF_K  + 128 * QP * 2)
#define OFF_AM  (OFF_KT + 128 * QP * 2)
#define OFF_VT  (OFF_AM + 128 * QP * 2)
#define OFF_ST  (OFF_VT + 64 * VP * 2)
#define OFF_Z   (OFF_ST + 64 * VP * 2)   // float[2][128]
#define OFF_D   (OFF_Z  + 2 * 128 * 4)   // float[128]
#define SMEM_BYTES (OFF_D + 128 * 4)     // = 175616 B (< 320 KB WGP LDS)

union UA  { v16bf v; u32x4 u[2]; };
union V8F { v8f   v; float f[8]; };

// A-operand fragment (16x32 bf16, row-major source).
// ISA layout: lanes 0-15 hold K={0..7,16..23} of row M=lane, lanes 16-31 hold
// K={8..15,24..31} of row M=lane-16.
__device__ __forceinline__ v16bf ld_a(const __bf16* tile, int pitch, int lane) {
  const int row = lane & 15;
  const int ko  = (lane >> 4) << 3;
  const __bf16* p = tile + row * pitch;
  UA r;
  r.u[0] = *(const u32x4*)(p + ko);
  r.u[1] = *(const u32x4*)(p + 16 + ko);
  return r.v;
}

// B-operand fragment (32x16 bf16) from a column-major (transposed) store:
// lane holds K={0..7,16..23} (lanes 0-15) or K={8..15,24..31} (lanes 16-31)
// of column n=lane%16.
__device__ __forceinline__ v16bf ld_b(const __bf16* tileT, int pitch, int lane) {
  const int col = lane & 15;
  const int ko  = (lane >> 4) << 3;
  const __bf16* p = tileT + col * pitch + ko;
  UA r;
  r.u[0] = *(const u32x4*)(p);
  r.u[1] = *(const u32x4*)(p + 16);
  return r.v;
}

__device__ __forceinline__ v8f wmma_bf16(v16bf a, v16bf b, v8f c) {
  return __builtin_amdgcn_wmma_f32_16x16x32_bf16(
      /*neg_a=*/false, a, /*neg_b=*/false, b,
      /*c_mod=*/(short)0, c, /*reuse_a=*/false, /*reuse_b=*/false);
}

// f32 global chunk (128x128) -> bf16 LDS, plus a transposed copy of K.
__device__ __forceinline__ void load_qk_chunk(const float* __restrict__ gq,
                                              const float* __restrict__ gk,
                                              __bf16* sQ, __bf16* sK,
                                              __bf16* sKT, int tid) {
  const f32x4* q4p = (const f32x4*)gq;
  const f32x4* k4p = (const f32x4*)gk;
#pragma unroll
  for (int j = 0; j < 16; ++j) {
    const int i4  = tid + j * 256;       // 4096 float4 per matrix
    const int row = i4 >> 5;             // 32 float4 per 128-wide row
    const int col = (i4 & 31) << 2;
    f32x4 q4 = q4p[i4];
    f32x4 k4 = k4p[i4];
    union { __bf16 h[4]; u32x2 d; } tq, tk;
    tq.h[0] = (__bf16)q4.x; tq.h[1] = (__bf16)q4.y;
    tq.h[2] = (__bf16)q4.z; tq.h[3] = (__bf16)q4.w;
    tk.h[0] = (__bf16)k4.x; tk.h[1] = (__bf16)k4.y;
    tk.h[2] = (__bf16)k4.z; tk.h[3] = (__bf16)k4.w;
    *(u32x2*)(sQ + row * QP + col) = tq.d;
    *(u32x2*)(sK + row * QP + col) = tk.d;
    sKT[(col + 0) * QP + row] = tk.h[0];
    sKT[(col + 1) * QP + row] = tk.h[1];
    sKT[(col + 2) * QP + row] = tk.h[2];
    sKT[(col + 3) * QP + row] = tk.h[3];
  }
}

__global__ __launch_bounds__(256, 1)
void linear_attn_chunk_kernel(const float* __restrict__ q,
                              const float* __restrict__ k,
                              const float* __restrict__ qr,
                              const float* __restrict__ kr,
                              const float* __restrict__ vv,
                              float* __restrict__ out) {
  extern __shared__ char smem[];
  __bf16* sQ  = (__bf16*)(smem + OFF_Q);
  __bf16* sK  = (__bf16*)(smem + OFF_K);
  __bf16* sKT = (__bf16*)(smem + OFF_KT);
  __bf16* sAm = (__bf16*)(smem + OFF_AM);
  __bf16* sVt = (__bf16*)(smem + OFF_VT);
  __bf16* sSt = (__bf16*)(smem + OFF_ST);
  float*  sZ  = (float*)(smem + OFF_Z);   // [2][128]
  float*  sD  = (float*)(smem + OFF_D);   // [128]

  const int tid  = threadIdx.x;
  const int lane = tid & 31;
  const int w    = tid >> 5;      // wave 0..7: owns token-strip w, d-strip w
  const int half = lane >> 4;
  const int nloc = lane & 15;

  const int bh = blockIdx.x;
  const float* gq  = q  + (size_t)bh * N_ * D_;
  const float* gk  = k  + (size_t)bh * N_ * D_;
  const float* gqr = qr + (size_t)bh * N_ * D_;
  const float* gkr = kr + (size_t)bh * N_ * D_;
  const float* gv  = vv + (size_t)bh * N_ * E_;
  float*       go  = out + (size_t)bh * N_ * E_;

  const v8f zero8 = {0.f, 0.f, 0.f, 0.f, 0.f, 0.f, 0.f, 0.f};

  V8F Sacc[2][4];                  // persistent f32 state: S and S_rot strips
#pragma unroll
  for (int br = 0; br < 2; ++br)
#pragma unroll
    for (int t = 0; t < 4; ++t) Sacc[br][t].v = zero8;

  if (tid < 128) { sZ[tid] = 0.f; sZ[128 + tid] = 0.f; sD[tid] = 0.f; }
  __syncthreads();

  for (int c = 0; c < NC; ++c) {
    // ---- stage V^T (f32 -> bf16, transposed: [e][m]) -----------------------
    {
      const f32x4* v4p = (const f32x4*)(gv + (size_t)c * CHUNK * E_);
#pragma unroll
      for (int j = 0; j < 8; ++j) {
        const int i4  = tid + j * 256;     // 2048 float4
        const int row = i4 >> 4;           // 16 float4 per 64-wide row
        const int col = (i4 & 15) << 2;
        f32x4 v4 = v4p[i4];
        sVt[(col + 0) * VP + row] = (__bf16)v4.x;
        sVt[(col + 1) * VP + row] = (__bf16)v4.y;
        sVt[(col + 2) * VP + row] = (__bf16)v4.z;
        sVt[(col + 3) * VP + row] = (__bf16)v4.w;
      }
    }
    if (c + 1 < NC) {                      // global_prefetch next chunk
      __builtin_prefetch(gq + (size_t)(c + 1) * CHUNK * D_, 0, 1);
      __builtin_prefetch(gk + (size_t)(c + 1) * CHUNK * D_, 0, 1);
    }

    V8F outacc[4];
#pragma unroll
    for (int t = 0; t < 4; ++t) outacc[t].v = zero8;

    for (int br = 0; br < 2; ++br) {
      const float* Qg = (br ? gqr : gq) + (size_t)c * CHUNK * D_;
      const float* Kg = (br ? gkr : gk) + (size_t)c * CHUNK * D_;

      // ---- publish this branch's state as bf16 S^T: [e][d] -----------------
#pragma unroll
      for (int t = 0; t < 4; ++t)
#pragma unroll
        for (int r = 0; r < 8; ++r) {
          const int e = 16 * t + nloc;
          const int d = 16 * w + r + 8 * half;
          sSt[e * VP + d] = (__bf16)Sacc[br][t].f[r];
        }

      load_qk_chunk(Qg, Kg, sQ, sK, sKT, tid);
      __syncthreads();

      // ---- A = tril(Qc Kc^T), strip w; rowsum -> D; bf16 mask -> LDS -------
      float rowD[8];
#pragma unroll
      for (int r = 0; r < 8; ++r) rowD[r] = 0.f;

      for (int ct = 0; ct < 8; ++ct) {
        V8F a; a.v = zero8;
        if (ct <= w) {
#pragma unroll
          for (int kk = 0; kk < 128; kk += 32)
            a.v = wmma_bf16(ld_a(sQ + 16 * w * QP + kk, QP, lane),
                            ld_b(sK + 16 * ct * QP + kk, QP, lane), a.v);
          if (ct == w) {                    // causal mask (inclusive diagonal)
#pragma unroll
            for (int r = 0; r < 8; ++r)
              if (nloc > r + 8 * half) a.f[r] = 0.f;
          }
        }
#pragma unroll
        for (int r = 0; r < 8; ++r) {
          rowD[r] += a.f[r];
          sAm[(16 * w + r + 8 * half) * QP + 16 * ct + nloc] = (__bf16)a.f[r];
        }
      }
      // reduce rowsums across the 16 lanes of each half-wave
#pragma unroll
      for (int r = 0; r < 8; ++r) {
        float s = rowD[r];
        s += __shfl_xor(s, 1, 32);
        s += __shfl_xor(s, 2, 32);
        s += __shfl_xor(s, 4, 32);
        s += __shfl_xor(s, 8, 32);
        rowD[r] = s;
      }
      if (nloc == 0) {
#pragma unroll
        for (int r = 0; r < 8; ++r)
          atomicAdd(&sD[16 * w + r + 8 * half], rowD[r]);
      }

      // ---- D += q . Z (pre-update Z) ---------------------------------------
      if (lane < 16) {
        const int n = 16 * w + lane;
        const __bf16* qp = sQ + n * QP;
        const float*  Zp = sZ + br * 128;
        float s = 0.f;
#pragma unroll 8
        for (int d = 0; d < 128; ++d) s += (float)qp[d] * Zp[d];
        atomicAdd(&sD[n], s);
      }

      // ---- out += Qc @ S  (inter-chunk) ------------------------------------
#pragma unroll
      for (int t = 0; t < 4; ++t)
#pragma unroll
        for (int kk = 0; kk < 128; kk += 32)
          outacc[t].v = wmma_bf16(ld_a(sQ + 16 * w * QP + kk, QP, lane),
                                  ld_b(sSt + 16 * t * VP + kk, VP, lane),
                                  outacc[t].v);

      // ---- out += tril(A) @ Vc  (intra-chunk; skip all-zero upper tiles) ---
#pragma unroll
      for (int t = 0; t < 4; ++t)
        for (int ki = 0; ki <= (w >> 1); ++ki) {
          const int kk = 32 * ki;
          outacc[t].v = wmma_bf16(ld_a(sAm + 16 * w * QP + kk, QP, lane),
                                  ld_b(sVt + 16 * t * VP + kk, VP, lane),
                                  outacc[t].v);
        }

      // ---- S += Kc^T @ Vc ---------------------------------------------------
#pragma unroll
      for (int t = 0; t < 4; ++t)
#pragma unroll
        for (int kk = 0; kk < 128; kk += 32)
          Sacc[br][t].v = wmma_bf16(ld_a(sKT + 16 * w * QP + kk, QP, lane),
                                    ld_b(sVt + 16 * t * VP + kk, VP, lane),
                                    Sacc[br][t].v);

      __syncthreads();
      // ---- Z += colsum(Kc) --------------------------------------------------
      if (lane < 16) {
        const int d = 16 * w + lane;
        float s = 0.f;
#pragma unroll 8
        for (int m = 0; m < 128; ++m) s += (float)sK[m * QP + d];
        sZ[br * 128 + d] += s;
      }
      __syncthreads();
    } // branch

    // ---- out[n,e] = acc * 1/(D + eps); store f32 ---------------------------
    float inv[8];
#pragma unroll
    for (int r = 0; r < 8; ++r) {
      const float dv = sD[16 * w + 8 * half + r];
      inv[r] = 1.f / (dv + 1e-12f);
    }
#pragma unroll
    for (int t = 0; t < 4; ++t)
#pragma unroll
      for (int r = 0; r < 8; ++r) {
        const int n = c * CHUNK + 16 * w + r + 8 * half;
        const int e = 16 * t + nloc;
        go[(size_t)n * E_ + e] = outacc[t].f[r] * inv[r];
      }
    __syncthreads();
    if (tid < 128) sD[tid] = 0.f;   // next chunk's normalizer accumulator
  } // chunk
}

extern "C" void kernel_launch(void* const* d_in, const int* in_sizes, int n_in,
                              void* d_out, int out_size, void* d_ws, size_t ws_size,
                              hipStream_t stream) {
  (void)in_sizes; (void)n_in; (void)out_size; (void)d_ws; (void)ws_size;
  const float* q  = (const float*)d_in[0];
  const float* k  = (const float*)d_in[1];
  const float* qr = (const float*)d_in[2];
  const float* kr = (const float*)d_in[3];
  const float* vv = (const float*)d_in[4];
  float* out = (float*)d_out;
  dim3 grid(B_ * H_), block(256);
  hipLaunchKernelGGL(linear_attn_chunk_kernel, grid, block, SMEM_BYTES, stream,
                     q, k, qr, kr, vv, out);
}
